// SparseStateSpaceModule_28750511079793
// MI455X (gfx1250) — compile-verified, matrix operand order checked by reading, not run on hardware
//
#include <hip/hip_runtime.h>
#include <hip/hip_bf16.h>

typedef __attribute__((ext_vector_type(16))) _Float16 v16h;
typedef __attribute__((ext_vector_type(8)))  float    v8f;

#define HW   16384
#define NB   128     // folded batches (B*HEADS*FOLD*FOLD)
#define NPIX 4096    // 64*64 pixels per folded batch
#define NM   64      // clusters

__device__ __forceinline__ float sigmoidf_(float x){ return 1.f/(1.f+__expf(-x)); }
__device__ __forceinline__ float siluf_(float x){ return x*sigmoidf_(x); }

#define WMMA(a,b,c) __builtin_amdgcn_wmma_f32_16x16x32_f16(false,(a),false,(b),(short)0,(c),false,false)

// A/B fragment loader. CDNA5 16-bit operand layout:
// lane L -> row (L&15); half i -> k = k0 + (i&7) + (i>>3)*16 + (L>>4)*8. k>=kmax -> 0 pad.
__device__ __forceinline__ v16h frag_f32(const float* p, long rowbase, int rstride,
                                         int kstride, int k0, int kmax, int lane){
  v16h f;
  long r = rowbase + (long)(lane & 15) * rstride;
  int kb = k0 + ((lane >> 4) << 3);
#pragma unroll
  for (int i = 0; i < 16; ++i){
    int k = kb + (i & 7) + ((i >> 3) << 4);
    f[i] = (k < kmax) ? (_Float16)p[r + (long)k * kstride] : (_Float16)0.f;
  }
  return f;
}
__device__ __forceinline__ v16h frag_f16(const _Float16* p, long rowbase, int rstride,
                                         int kstride, int k0, int kmax, int lane){
  v16h f;
  long r = rowbase + (long)(lane & 15) * rstride;
  int kb = k0 + ((lane >> 4) << 3);
#pragma unroll
  for (int i = 0; i < 16; ++i){
    int k = kb + (i & 7) + ((i >> 3) << 4);
    f[i] = (k < kmax) ? p[r + (long)k * kstride] : (_Float16)0.f;
  }
  return f;
}

// ---------- K0: one-shot weight conversion f32 -> f16
__global__ void k0_cvt(const float* __restrict__ src, _Float16* __restrict__ dst, int n){
  int i = blockIdx.x*256 + threadIdx.x;
  if (i < n) dst[i] = (_Float16)src[i];
}

// ---------- K1: in_proj GEMM (96->96), split into xi (NCHW) and silu(z) (token-major)
__global__ void k1_inproj(const float* __restrict__ x, const _Float16* __restrict__ wh,
                          float* __restrict__ xi, float* __restrict__ zs){
  int tile = blockIdx.x, lane = threadIdx.x & 31;
  int b = tile >> 10;            // 1024 tiles of 16 tokens per batch image
  int hw0 = (tile & 1023) << 4;
  long arow = (long)b*96*HW + hw0;     // NCHW: k-stride = HW, token-stride = 1
  v16h a[3];
#pragma unroll
  for (int ks = 0; ks < 3; ++ks) a[ks] = frag_f32(x, arow, 1, HW, ks*32, 96, lane);
  for (int nt = 0; nt < 6; ++nt){
    v8f acc = {};
#pragma unroll
    for (int ks = 0; ks < 3; ++ks)
      acc = WMMA(a[ks], frag_f16(wh, (long)nt*16*96, 96, 1, ks*32, 96, lane), acc);
    int n  = nt*16 + (lane & 15);
    int mb = (lane >> 4) << 3;
#pragma unroll
    for (int r = 0; r < 8; ++r){
      int tok = mb + r;
      float v = acc[r];
      if (n < 48) xi[(long)b*48*HW + (long)n*HW + hw0 + tok] = v;
      else        zs[((long)b*HW + hw0 + tok)*48 + (n-48)] = siluf_(v);
    }
  }
}

// ---------- K2: depthwise 3x3 conv + bias + silu
__global__ void k2_dwconv(const float* __restrict__ xi, const float* __restrict__ cw,
                          const float* __restrict__ cb, float* __restrict__ xc){
  long gid = (long)blockIdx.x*256 + threadIdx.x;
  int w_ = gid & 127, h_ = (gid >> 7) & 127;
  int c  = (gid >> 14) % 48;
  int b  = (int)(gid / ((long)48*HW));
  float s = cb[c];
  const float* base = xi + ((long)b*48 + c)*HW;
#pragma unroll
  for (int kh = 0; kh < 3; ++kh){
    int hh = h_ + kh - 1; if ((unsigned)hh >= 128u) continue;
#pragma unroll
    for (int kw = 0; kw < 3; ++kw){
      int ww = w_ + kw - 1; if ((unsigned)ww >= 128u) continue;
      s += base[hh*128 + ww] * cw[c*9 + kh*3 + kw];
    }
  }
  xc[gid] = siluf_(s);
}

// ---------- K3: f/value 1x1 conv GEMM (48 -> 384), scatter into folded pixel layout (f16)
__global__ void k3_fv(const float* __restrict__ xc,
                      const _Float16* __restrict__ f_wh, const float* __restrict__ f_b,
                      const _Float16* __restrict__ v_wh, const float* __restrict__ v_b,
                      _Float16* __restrict__ xf_t, _Float16* __restrict__ val_t){
  int tile = blockIdx.x, lane = threadIdx.x & 31;
  int b = tile >> 10;
  int hw0 = (tile & 1023) << 4;
  int h_ = hw0 >> 7, w0 = hw0 & 127;
  long arow = (long)b*48*HW + hw0;
  v16h a0 = frag_f32(xc, arow, 1, HW, 0, 48, lane);
  v16h a1 = frag_f32(xc, arow, 1, HW, 32, 48, lane);
  for (int nt = 0; nt < 24; ++nt){
    int n0 = nt*16;
    const _Float16* wp = (n0 < 192) ? f_wh : v_wh;
    int nb = (n0 < 192) ? n0 : n0 - 192;
    v8f acc = {};
    acc = WMMA(a0, frag_f16(wp, (long)nb*48, 48, 1, 0, 48, lane), acc);
    acc = WMMA(a1, frag_f16(wp, (long)nb*48, 48, 1, 32, 48, lane), acc);
    int nl = nb + (lane & 15);
    float bias = (n0 < 192) ? f_b[nl] : v_b[nl];
    int hd = nl / 48, c = nl % 48;
    int mb = (lane >> 4) << 3;
#pragma unroll
    for (int r = 0; r < 8; ++r){
      int w_ = w0 + mb + r;
      int f1 = h_ >> 6, w2 = h_ & 63, f2 = w_ >> 6, h2 = w_ & 63;
      int bb = ((b*4 + hd)*2 + f1)*2 + f2;
      long pix = (long)bb*NPIX + w2*64 + h2;
      float v = acc[r] + bias;
      if (n0 < 192) xf_t[pix*48 + c]  = (_Float16)v;
      else          val_t[pix*48 + c] = (_Float16)v;
    }
  }
}

// ---------- K4a: 8x8 pooling -> centers (l2-normalized, f16) and value_centers (f32)
__global__ void k4a_pool(const _Float16* __restrict__ xf_t, const _Float16* __restrict__ val_t,
                         _Float16* __restrict__ cxn, float* __restrict__ vcent){
  int gid = blockIdx.x;            // NB*NM
  int bb = gid >> 6, m = gid & 63;
  int p1 = m >> 3, p2 = m & 7;
  int tid = threadIdx.x;           // 64
  __shared__ float cs[48];
  float sc = 0.f, sv = 0.f;
  if (tid < 48){
    for (int i = 0; i < 8; ++i)
      for (int j = 0; j < 8; ++j){
        long pix = (long)bb*NPIX + (p1*8+i)*64 + (p2*8+j);
        sc += (float)xf_t[pix*48 + tid];
        sv += (float)val_t[pix*48 + tid];
      }
    sc *= (1.f/64.f); sv *= (1.f/64.f);
    cs[tid] = sc;
  }
  __syncthreads();
  if (tid < 48){
    float nrm = 0.f;
    for (int d = 0; d < 48; ++d) nrm += cs[d]*cs[d];
    nrm = fmaxf(sqrtf(nrm), 1e-12f);
    cxn[((long)bb*NM + m)*48 + tid]  = (_Float16)(sc / nrm);
    vcent[((long)bb*NM + m)*48 + tid] = sv;
  }
}

// ---------- K4b: l2-normalize pixel features in place (xf_t -> xn)
__global__ void k4b_norm(_Float16* __restrict__ xf_t){
  long pid = (long)blockIdx.x*256 + threadIdx.x;
  _Float16* row = xf_t + pid*48;
  float n = 0.f;
  for (int d = 0; d < 48; ++d){ float v = (float)row[d]; n += v*v; }
  float inv = 1.f / fmaxf(sqrtf(n), 1e-12f);
  for (int d = 0; d < 48; ++d) row[d] = (_Float16)((float)row[d]*inv);
}

// ---------- K5: sim logits via WMMA (64 centers x pixels), fused argmax+sigmoid
__global__ void k5_sim(const _Float16* __restrict__ xn, const _Float16* __restrict__ cxn,
                       const float* __restrict__ alpha, const float* __restrict__ beta,
                       int* __restrict__ simi, float* __restrict__ simv){
  int blk = blockIdx.x;                    // NB * 64
  int bb = blk >> 6;
  int wv = threadIdx.x >> 5, lane = threadIdx.x & 31;
  int n0 = (blk & 63)*64 + wv*16;          // 16 pixels per wave
  long brow = ((long)bb*NPIX + n0)*48;
  v16h b0 = frag_f16(xn, brow, 48, 1, 0, 48, lane);
  v16h b1 = frag_f16(xn, brow, 48, 1, 32, 48, lane);
  float bestv = -1e30f; int bestm = 0;
  for (int mt = 0; mt < 4; ++mt){
    long arow = ((long)bb*NM + mt*16)*48;
    v8f acc = {};
    acc = WMMA(frag_f16(cxn, arow, 48, 1, 0, 48, lane), b0, acc);
    acc = WMMA(frag_f16(cxn, arow, 48, 1, 32, 48, lane), b1, acc);
    int mb = mt*16 + ((lane >> 4) << 3);
#pragma unroll
    for (int r = 0; r < 8; ++r){
      float v = acc[r];
      if (v > bestv){ bestv = v; bestm = mb + r; }
    }
  }
  float ov = __shfl_xor(bestv, 16, 32);    // combine the two half-wave M groups
  int   om = __shfl_xor(bestm, 16, 32);
  if (ov > bestv || (ov == bestv && om < bestm)){ bestv = ov; bestm = om; }
  if (lane < 16){
    long n = (long)bb*NPIX + n0 + lane;
    simi[n] = bestm;
    simv[n] = sigmoidf_(beta[0] + alpha[0]*bestv);
  }
}

// ---------- K6: sparse cluster aggregation (LDS atomics) -> xs (bb, d, l)
__global__ void k6_agg(const int* __restrict__ simi, const float* __restrict__ simv,
                       const _Float16* __restrict__ val_t, const float* __restrict__ vcent,
                       float* __restrict__ xs){
  int bb = blockIdx.x, tid = threadIdx.x;  // 256
  __shared__ float acc[NM*48];
  __shared__ float den[NM];
  for (int i = tid; i < NM*48; i += 256) acc[i] = 0.f;
  if (tid < NM) den[tid] = 0.f;
  __syncthreads();
  for (int p = tid; p < NPIX; p += 256){
    int m = simi[(long)bb*NPIX + p];
    float s = simv[(long)bb*NPIX + p];
    atomicAdd(&den[m], s);
    const _Float16* vr = val_t + ((long)bb*NPIX + p)*48;
    for (int c = 0; c < 48; ++c) atomicAdd(&acc[m*48 + c], s*(float)vr[c]);
  }
  __syncthreads();
  for (int i = tid; i < NM*48; i += 256){
    int m = i / 48, c = i % 48;
    float o = (acc[i] + vcent[((long)bb*NM + m)*48 + c]) / (den[m] + 1.f);
    xs[((long)bb*48 + c)*64 + m] = o;
  }
}

// ---------- K7: x_proj + depthwise conv1d(k=7) + dt projection + softplus
__global__ void k7_xproj(const float* __restrict__ xs, const float* __restrict__ xpw,
                         const float* __restrict__ xcw, const float* __restrict__ xcb,
                         const float* __restrict__ dtw, const float* __restrict__ dtb,
                         float* __restrict__ delta, float* __restrict__ Bs, float* __restrict__ Cs){
  int bb = blockIdx.x, l = threadIdx.x;    // 64 threads = sequence positions
  __shared__ float d1[22*64];
  __shared__ float d2[22*64];
  for (int r = 0; r < 22; ++r){
    float s = 0.f;
    for (int c = 0; c < 48; ++c) s += xpw[r*48 + c]*xs[((long)bb*48 + c)*64 + l];
    d1[r*64 + l] = s;
  }
  __syncthreads();
  for (int r = 0; r < 22; ++r){
    float s = xcb[r];
    for (int t = 0; t < 7; ++t){
      int ll = l + t - 3;
      if ((unsigned)ll < 64u) s += d1[r*64 + ll]*xcw[r*7 + t];
    }
    d2[r*64 + l] = s;
  }
  __syncthreads();
  for (int d = 0; d < 48; ++d){
    float s = dtb[d];
    for (int r = 0; r < 6; ++r) s += dtw[d*6 + r]*d2[r*64 + l];
    float dl = (s > 20.f) ? s : log1pf(__expf(s));
    delta[((long)bb*48 + d)*64 + l] = dl;
  }
  for (int n = 0; n < 8; ++n){
    Bs[((long)bb*8 + n)*64 + l] = d2[(6+n)*64 + l];
    Cs[((long)bb*8 + n)*64 + l] = d2[(14+n)*64 + l];
  }
}

// ---------- K8: selective scan, thread per (bb, d)
__global__ void k8_scan(const float* __restrict__ xs, const float* __restrict__ delta,
                        const float* __restrict__ Bs, const float* __restrict__ Cs,
                        const float* __restrict__ A_logs, const float* __restrict__ Ds,
                        float* __restrict__ yv){
  int gid = blockIdx.x*128 + threadIdx.x;
  if (gid >= NB*48) return;
  int bb = gid / 48, d = gid % 48;
  float A[8], h[8];
#pragma unroll
  for (int n = 0; n < 8; ++n){ A[n] = -__expf(A_logs[d*8 + n]); h[n] = 0.f; }
  float Dd = Ds[d];
  for (int l = 0; l < 64; ++l){
    float dl = delta[((long)bb*48 + d)*64 + l];
    float u  = xs[((long)bb*48 + d)*64 + l];
    float y = 0.f;
#pragma unroll
    for (int n = 0; n < 8; ++n){
      h[n] = __expf(dl*A[n])*h[n] + dl*u*Bs[((long)bb*8 + n)*64 + l];
      y += h[n]*Cs[((long)bb*8 + n)*64 + l];
    }
    yv[((long)bb*64 + l)*48 + d] = y + Dd*u;
  }
}

// ---------- K9a: scatter scan output back to pixels (sparse sim), unfold -> token vectors (f16)
__global__ void k9a_gather(const int* __restrict__ simi, const float* __restrict__ simv,
                           const float* __restrict__ yv, _Float16* __restrict__ tokvec){
  long gid = (long)blockIdx.x*256 + threadIdx.x;   // 131072 tokens * 4 heads
  int hd = gid & 3;
  long tok = gid >> 2;
  int w_ = tok & 127, h_ = (tok >> 7) & 127, b = (int)(tok >> 14);
  int f1 = h_ >> 6, w2 = h_ & 63, f2 = w_ >> 6, h2 = w_ & 63;
  int bb = ((b*4 + hd)*2 + f1)*2 + f2;
  long pix = (long)bb*NPIX + w2*64 + h2;
  int m = simi[pix]; float s = simv[pix];
  const float* yr = yv + ((long)bb*NM + m)*48;
  _Float16* tr = tokvec + tok*192 + hd*48;
  for (int c = 0; c < 48; ++c) tr[c] = (_Float16)(yr[c]*s);
}

// ---------- K9b: proj GEMM (192 -> 48) -> yc (NCHW); A tile staged via LDS to avoid spills
__global__ void k9b_proj(const _Float16* __restrict__ tokvec, const _Float16* __restrict__ pwh,
                         const float* __restrict__ pb, float* __restrict__ yc){
  int tile = blockIdx.x, lane = threadIdx.x & 31;
  int b = tile >> 10;
  int hw0 = (tile & 1023) << 4;
  __shared__ _Float16 at[16*192];
  {
    const unsigned* src = (const unsigned*)(tokvec + ((long)b*HW + hw0)*192);
    unsigned* dst = (unsigned*)at;
    for (int i = lane; i < 1536; i += 32) dst[i] = src[i];
  }
  __syncthreads();
  for (int nt = 0; nt < 3; ++nt){
    v8f acc = {};
    for (int ks = 0; ks < 6; ++ks){
      v16h a  = frag_f16(at, 0, 192, 1, ks*32, 192, lane);
      v16h bf = frag_f16(pwh, (long)nt*16*192, 192, 1, ks*32, 192, lane);
      acc = WMMA(a, bf, acc);
    }
    int n  = nt*16 + (lane & 15);
    float bias = pb[n];
    int mb = (lane >> 4) << 3;
#pragma unroll
    for (int r = 0; r < 8; ++r)
      yc[((long)b*48 + n)*HW + hw0 + mb + r] = acc[r] + bias;
  }
}

// ---------- K9c: regroup (c,w)-flatten, LayerNorm, silu(z) gate, WMMA out_proj (48->96)
__global__ void k9c_tail(const float* __restrict__ yc, const float* __restrict__ zs,
                         const float* __restrict__ onw, const float* __restrict__ onb,
                         const _Float16* __restrict__ opwh, float* __restrict__ out){
  int gid = blockIdx.x;          // B*H = 1024
  int b = gid >> 7, h = gid & 127;
  int tid = threadIdx.x;         // 128 = 4 waves
  __shared__ float    ly[128*48];
  __shared__ _Float16 lh[128*48];
  for (int i = tid; i < 6144; i += 128){
    int c = i >> 7, w = i & 127;     // flatten (c,w) -> rows of 48 (matches reference reshape)
    ly[i] = yc[(((long)b*48 + c)*128 + h)*128 + w];
  }
  __syncthreads();
  {
    int j = tid;
    float mu = 0.f;
    for (int d = 0; d < 48; ++d) mu += ly[j*48 + d];
    mu *= (1.f/48.f);
    float var = 0.f;
    for (int d = 0; d < 48; ++d){ float t = ly[j*48 + d] - mu; var += t*t; }
    var *= (1.f/48.f);
    float inv = rsqrtf(var + 1e-5f);
    const float* zr = zs + (((long)b*128 + h)*128 + j)*48;
    for (int d = 0; d < 48; ++d){
      float nv = (ly[j*48 + d] - mu)*inv*onw[d] + onb[d];
      lh[j*48 + d] = (_Float16)(nv * zr[d]);
    }
  }
  __syncthreads();
  int wv = tid >> 5, lane = tid & 31;
  const _Float16* lp = lh;
  for (int mi = 0; mi < 2; ++mi){
    int mt = wv*2 + mi;
    v16h a0 = frag_f16(lp, (long)mt*16*48, 48, 1, 0, 48, lane);
    v16h a1 = frag_f16(lp, (long)mt*16*48, 48, 1, 32, 48, lane);
    for (int nt = 0; nt < 6; ++nt){
      v8f acc = {};
      acc = WMMA(a0, frag_f16(opwh, (long)nt*16*48, 48, 1, 0, 48, lane), acc);
      acc = WMMA(a1, frag_f16(opwh, (long)nt*16*48, 48, 1, 32, 48, lane), acc);
      int o  = nt*16 + (lane & 15);
      int mb = (lane >> 4) << 3;
#pragma unroll
      for (int r = 0; r < 8; ++r){
        int w = mt*16 + mb + r;
        out[(((long)b*96 + o)*128 + h)*128 + w] = acc[r];
      }
    }
  }
}

extern "C" void kernel_launch(void* const* d_in, const int* in_sizes, int n_in,
                              void* d_out, int out_size, void* d_ws, size_t ws_size,
                              hipStream_t stream){
  (void)in_sizes; (void)n_in; (void)out_size; (void)ws_size;
  const float* x        = (const float*)d_in[0];
  const float* in_w     = (const float*)d_in[1];
  const float* c2w      = (const float*)d_in[2];
  const float* c2b      = (const float*)d_in[3];
  const float* f_w      = (const float*)d_in[4];
  const float* f_b      = (const float*)d_in[5];
  const float* v_w      = (const float*)d_in[6];
  const float* v_b      = (const float*)d_in[7];
  const float* proj_w   = (const float*)d_in[8];
  const float* proj_b   = (const float*)d_in[9];
  const float* xpw      = (const float*)d_in[10];
  const float* xcw      = (const float*)d_in[11];
  const float* xcb      = (const float*)d_in[12];
  const float* dtw      = (const float*)d_in[13];
  const float* dtb      = (const float*)d_in[14];
  const float* A_logs   = (const float*)d_in[15];
  const float* Ds       = (const float*)d_in[16];
  const float* onw      = (const float*)d_in[17];
  const float* onb      = (const float*)d_in[18];
  const float* opw      = (const float*)d_in[19];
  const float* sim_a    = (const float*)d_in[20];
  const float* sim_b    = (const float*)d_in[21];
  float* out = (float*)d_out;

  char* ws = (char*)d_ws;
  size_t off = 0;
  auto alloc = [&](size_t bytes)->char*{
    char* p = ws + off; off += (bytes + 255) & ~(size_t)255; return p;
  };
  float*    zs    = (float*)   alloc((size_t)131072*48*4);        // silu(z), token-major
  float*    xi    = (float*)   alloc((size_t)8*48*HW*4);          // conv input; reused as yc
  float*    xc    = (float*)   alloc((size_t)8*48*HW*4);          // silu(dwconv)
  _Float16* xf    = (_Float16*)alloc((size_t)NB*NPIX*48*2);       // raw f -> normalized xn -> reused as tokvec
  _Float16* vt    = (_Float16*)alloc((size_t)NB*NPIX*48*2);       // value per pixel
  _Float16* cxn   = (_Float16*)alloc((size_t)NB*NM*48*2);         // normalized centers
  float*    vcent = (float*)   alloc((size_t)NB*NM*48*4);
  int*      simi  = (int*)     alloc((size_t)NB*NPIX*4);
  float*    simv  = (float*)   alloc((size_t)NB*NPIX*4);
  float*    xs    = (float*)   alloc((size_t)NB*48*64*4);
  float*    dlt   = (float*)   alloc((size_t)NB*48*64*4);
  float*    Bss   = (float*)   alloc((size_t)NB*8*64*4);
  float*    Css   = (float*)   alloc((size_t)NB*8*64*4);
  float*    yv    = (float*)   alloc((size_t)NB*64*48*4);
  _Float16* in_wh = (_Float16*)alloc((size_t)96*96*2);            // f16 weight copies
  _Float16* f_wh  = (_Float16*)alloc((size_t)192*48*2);
  _Float16* v_wh  = (_Float16*)alloc((size_t)192*48*2);
  _Float16* p_wh  = (_Float16*)alloc((size_t)48*192*2);
  _Float16* o_wh  = (_Float16*)alloc((size_t)96*48*2);
  float*    yc    = xi;                   // reuse (xi dead after K2)
  _Float16* tokv  = xf;                   // reuse (xn dead after K5); same byte size

  k0_cvt    <<<36, 256, 0, stream>>>(in_w,   in_wh, 96*96);
  k0_cvt    <<<36, 256, 0, stream>>>(f_w,    f_wh,  192*48);
  k0_cvt    <<<36, 256, 0, stream>>>(v_w,    v_wh,  192*48);
  k0_cvt    <<<36, 256, 0, stream>>>(proj_w, p_wh,  48*192);
  k0_cvt    <<<18, 256, 0, stream>>>(opw,    o_wh,  96*48);

  k1_inproj <<<8192, 32, 0, stream>>>(x, in_wh, xi, zs);
  k2_dwconv <<<(8*48*HW)/256, 256, 0, stream>>>(xi, c2w, c2b, xc);
  k3_fv     <<<8192, 32, 0, stream>>>(xc, f_wh, f_b, v_wh, v_b, xf, vt);
  k4a_pool  <<<NB*NM, 64, 0, stream>>>(xf, vt, cxn, vcent);
  k4b_norm  <<<(NB*NPIX)/256, 256, 0, stream>>>(xf);
  k5_sim    <<<NB*64, 128, 0, stream>>>(xf, cxn, sim_a, sim_b, simi, simv);
  k6_agg    <<<NB, 256, 0, stream>>>(simi, simv, vt, vcent, xs);
  k7_xproj  <<<NB, 64, 0, stream>>>(xs, xpw, xcw, xcb, dtw, dtb, dlt, Bss, Css);
  k8_scan   <<<48, 128, 0, stream>>>(xs, dlt, Bss, Css, A_logs, Ds, yv);
  k9a_gather<<<2048, 256, 0, stream>>>(simi, simv, yv, tokv);
  k9b_proj  <<<8192, 32, 0, stream>>>(tokv, p_wh, proj_b, yc);
  k9c_tail  <<<1024, 128, 0, stream>>>(yc, zs, onw, onb, o_wh, out);
}
